// JacobAttentionLayer_84018150244754
// MI455X (gfx1250) — compile-verified, matrix operand order checked by reading, not run on hardware
//
#include <hip/hip_runtime.h>
#include <hip/hip_bf16.h>

typedef __bf16 bf16;
typedef __attribute__((ext_vector_type(16))) __bf16 v16bf;
typedef __attribute__((ext_vector_type(8)))  __bf16 v8bf;
typedef __attribute__((ext_vector_type(4)))  __bf16 v4bf;
typedef __attribute__((ext_vector_type(8)))  float  v8f;

constexpr int BB = 2;
constexpr int TT = 2048;
constexpr int CC = 1024;
constexpr int NHEAD = 16;
constexpr int HS = 64;
constexpr int OC = 3 * CC + NHEAD;   // 3088
constexpr int LDS_STRIDE = 72;       // 64 + 8 pad (bf16 elems), 144B rows

// Build a 16x32-bf16 WMMA A/B fragment for this lane from two 16B segments:
// lo -> K = kbase..kbase+7, hi -> K = kbase+16..kbase+23 (ISA 7.12.2 layout).
__device__ __forceinline__ v16bf make_frag(const bf16* lo, const bf16* hi) {
  v8bf a = *reinterpret_cast<const v8bf*>(lo);
  v8bf b = *reinterpret_cast<const v8bf*>(hi);
  v16bf r;
#pragma unroll
  for (int i = 0; i < 8; ++i) { r[i] = a[i]; r[8 + i] = b[i]; }
  return r;
}

// ---------------- fp32 -> bf16 conversion ----------------
__global__ void cvt_f32_bf16(const float* __restrict__ src, bf16* __restrict__ dst, int n4) {
  int i = blockIdx.x * blockDim.x + threadIdx.x;
  if (i < n4) {
    float4 v = reinterpret_cast<const float4*>(src)[i];
    v4bf o;
    o[0] = (bf16)v.x; o[1] = (bf16)v.y; o[2] = (bf16)v.z; o[3] = (bf16)v.w;
    reinterpret_cast<v4bf*>(dst)[i] = o;
  }
}

// ---------------- QKV+gate projection GEMM ----------------
// 4 waves/block; block tile 128(M) x 64(N), K-step 64; LDS double buffered.
// Wave w owns a 32-row M-strip (two 16-row WMMA strips) x all four N-subtiles:
// 16 WMMAs per K-step per wave, B fragments reused x2, A fragments reused x4.
__global__ __launch_bounds__(128) void qkv_gemm(
    const bf16* __restrict__ X, const bf16* __restrict__ Wb,
    bf16* __restrict__ Q, bf16* __restrict__ K, bf16* __restrict__ V,
    float* __restrict__ gates) {
  __shared__ bf16 ldsX[2][128 * LDS_STRIDE];
  __shared__ bf16 ldsW[2][64 * LDS_STRIDE];

  const int tid  = threadIdx.x;
  const int wave = tid >> 5;
  const int lane = tid & 31;
  const int half = (lane < 16) ? 0 : 8;
  const int ln   = lane & 15;
  const int n0   = blockIdx.x * 64;          // output col base (edge block clamps)
  const int m0   = blockIdx.y * 128;         // output row base

  // per-thread staging: X tile 128x64 -> 8 x 16B, W tile 64x64 -> 4 x 16B
  v8bf xr[8], wr[4];
  auto load_tiles = [&](int k0) {
#pragma unroll
    for (int p = 0; p < 8; ++p) {
      int idx = tid + p * 128, row = idx >> 3, seg = (idx & 7) * 8;
      xr[p] = *reinterpret_cast<const v8bf*>(X + (size_t)(m0 + row) * CC + k0 + seg);
    }
#pragma unroll
    for (int p = 0; p < 4; ++p) {
      int idx = tid + p * 128, row = idx >> 3, seg = (idx & 7) * 8;
      int wn = n0 + row; if (wn >= OC) wn = OC - 1;   // clamp edge reads
      wr[p] = *reinterpret_cast<const v8bf*>(Wb + (size_t)wn * CC + k0 + seg);
    }
  };
  auto store_tiles = [&](int buf) {
#pragma unroll
    for (int p = 0; p < 8; ++p) {
      int idx = tid + p * 128, row = idx >> 3, seg = (idx & 7) * 8;
      *reinterpret_cast<v8bf*>(&ldsX[buf][row * LDS_STRIDE + seg]) = xr[p];
    }
#pragma unroll
    for (int p = 0; p < 4; ++p) {
      int idx = tid + p * 128, row = idx >> 3, seg = (idx & 7) * 8;
      *reinterpret_cast<v8bf*>(&ldsW[buf][row * LDS_STRIDE + seg]) = wr[p];
    }
  };

  v8f acc[2][4] = {};
  load_tiles(0);
  int buf = 0;
  constexpr int KSTEPS = CC / 64;            // 16
  for (int kt = 0; kt < KSTEPS; ++kt) {
    store_tiles(buf);
    __syncthreads();
    if (kt + 1 < KSTEPS) load_tiles((kt + 1) * 64);   // overlap next loads with WMMAs
#pragma unroll
    for (int ks = 0; ks < 2; ++ks) {
      const bf16* ar0 = &ldsX[buf][(wave * 32 + ln) * LDS_STRIDE + ks * 32];
      const bf16* ar1 = &ldsX[buf][(wave * 32 + 16 + ln) * LDS_STRIDE + ks * 32];
      v16bf af0 = make_frag(ar0 + half, ar0 + half + 16);
      v16bf af1 = make_frag(ar1 + half, ar1 + half + 16);
#pragma unroll
      for (int j = 0; j < 4; ++j) {
        const bf16* brow = &ldsW[buf][(j * 16 + ln) * LDS_STRIDE + ks * 32];
        v16bf bfr = make_frag(brow + half, brow + half + 16);
        acc[0][j] = __builtin_amdgcn_wmma_f32_16x16x32_bf16(
            false, af0, false, bfr, (short)0, acc[0][j], false, false);
        acc[1][j] = __builtin_amdgcn_wmma_f32_16x16x32_bf16(
            false, af1, false, bfr, (short)0, acc[1][j], false, false);
      }
    }
    buf ^= 1;
  }

  // writeback: route columns to Q/K/V (bf16, [b][h][t][d]) or gate logits (f32)
#pragma unroll
  for (int s = 0; s < 2; ++s) {
#pragma unroll
    for (int j = 0; j < 4; ++j) {
      const int n = n0 + j * 16 + ln;
      if (n >= OC) continue;
#pragma unroll
      for (int r = 0; r < 8; ++r) {
        int m = m0 + wave * 32 + s * 16 + half + r;
        int b = m >> 11, t = m & (TT - 1);
        float val = acc[s][j][r];
        if (n < CC) {
          int hh = n >> 6, d = n & 63;
          Q[(((size_t)(b * NHEAD + hh)) * TT + t) * HS + d] = (bf16)val;
        } else if (n < 2 * CC) {
          int nn = n - CC; int hh = nn >> 6, d = nn & 63;
          K[(((size_t)(b * NHEAD + hh)) * TT + t) * HS + d] = (bf16)val;
        } else if (n < 3 * CC) {
          int nn = n - 2 * CC; int hh = nn >> 6, d = nn & 63;
          V[(((size_t)(b * NHEAD + hh)) * TT + t) * HS + d] = (bf16)val;
        } else {
          gates[((size_t)(b * TT + t)) * NHEAD + (n - 3 * CC)] = val;
        }
      }
    }
  }
}

// ---------------- gate log-sigmoid cumulative scan ----------------
__global__ void gate_scan(const float* __restrict__ gates, float* __restrict__ logG) {
  int i = threadIdx.x;                        // one lane per (b,h), 32 total
  if (i >= BB * NHEAD) return;
  int b = i >> 4, h = i & 15;
  float acc = 0.f;
  for (int t = 0; t < TT; ++t) {
    float g  = gates[((size_t)(b * TT + t)) * NHEAD + h];
    float ls = fminf(g, 0.f) - log1pf(__expf(-fabsf(g)));  // log_sigmoid
    acc += ls;
    logG[(size_t)i * TT + t] = acc;
  }
}

// ---------------- fused log-space flash attention ----------------
__global__ __launch_bounds__(128) void attn(
    const bf16* __restrict__ Q, const bf16* __restrict__ K, const bf16* __restrict__ V,
    const float* __restrict__ logG, float* __restrict__ out) {
  __shared__ bf16 ldsK [64 * LDS_STRIDE];      // K tile, row-major
  __shared__ bf16 ldsVt[64 * LDS_STRIDE];      // V tile, transposed [d][t]
  __shared__ bf16 ldsP [4][16 * LDS_STRIDE];   // per-wave P strip
  __shared__ float ldsGk[64];

  const int tid  = threadIdx.x;
  const int wave = tid >> 5;
  const int lane = tid & 31;
  const int half = (lane < 16) ? 0 : 8;
  const int ln   = lane & 15;

  const int qt = blockIdx.x;                   // query tile (64 rows)
  const int bh = blockIdx.y;                   // b*NHEAD + h
  const int b  = bh >> 4;
  const int h  = bh & 15;
  const int qstart = qt * 64;

  const bf16* Qh = Q + (size_t)bh * TT * HS;
  const bf16* Kh = K + (size_t)bh * TT * HS;
  const bf16* Vh = V + (size_t)bh * TT * HS;
  const float* G = logG + (size_t)bh * TT;

  // Q fragments (A-matrix, M=16 rows of this wave, K split 0..31 / 32..63)
  const bf16* qrow = Qh + (size_t)(qstart + wave * 16 + ln) * HS;
  v16bf qf0 = make_frag(qrow + half,      qrow + half + 16);
  v16bf qf1 = make_frag(qrow + 32 + half, qrow + 32 + half + 16);

  float gq[8], m_run[8], l_run[8];
  v8f o[4] = {};
#pragma unroll
  for (int r = 0; r < 8; ++r) {
    gq[r]    = G[qstart + wave * 16 + half + r];
    m_run[r] = -1e30f;
    l_run[r] = 0.f;
  }

  const float LOGSCALE = -2.0794415416798357f; // log(1/sqrt(64))
  const float EPS = 1e-7f;

  for (int kt = 0; kt <= qt; ++kt) {
    const int kvstart = kt * 64;
    const bool diag = (kt == qt);
    __syncthreads();                           // prior-iter LDS readers done
    // stage K tile (row-major, 16B vectors)
    for (int idx = tid; idx < 64 * 8; idx += 128) {
      int row = idx >> 3, seg = (idx & 7) * 8;
      *reinterpret_cast<v8bf*>(&ldsK[row * LDS_STRIDE + seg]) =
          *reinterpret_cast<const v8bf*>(Kh + (size_t)(kvstart + row) * HS + seg);
    }
    // stage V tile transposed: ldsVt[d][t]
    for (int idx = tid; idx < 64 * 8; idx += 128) {
      int row = idx >> 3, seg = (idx & 7) * 8;
      v8bf v = *reinterpret_cast<const v8bf*>(Vh + (size_t)(kvstart + row) * HS + seg);
#pragma unroll
      for (int e = 0; e < 8; ++e) ldsVt[(seg + e) * LDS_STRIDE + row] = v[e];
    }
    if (tid < 64) ldsGk[tid] = G[kvstart + tid];
    __syncthreads();

    // prefetch next K/V tiles into cache while we compute on this one
    if (kt + 1 <= qt) {
      const bf16* nk = Kh + (size_t)(kvstart + 64 + (tid & 63)) * HS;
      const bf16* nv = Vh + (size_t)(kvstart + 64 + (tid & 63)) * HS;
      __builtin_prefetch(nk, 0, 1);
      __builtin_prefetch(nv, 0, 1);
    }

    // S strip = Q(16x64) @ K_tile^T(64x64) -> 4 accumulators of 16x16
    v8f s[4];
#pragma unroll
    for (int j = 0; j < 4; ++j) {
      v8f a = {};
      const bf16* krow = &ldsK[(j * 16 + ln) * LDS_STRIDE];   // B[k][n]=Ktile[n][k]
      v16bf b0 = make_frag(krow + half,      krow + half + 16);
      a = __builtin_amdgcn_wmma_f32_16x16x32_bf16(false, qf0, false, b0, (short)0, a, false, false);
      v16bf b1 = make_frag(krow + 32 + half, krow + 32 + half + 16);
      a = __builtin_amdgcn_wmma_f32_16x16x32_bf16(false, qf1, false, b1, (short)0, a, false, false);
      s[j] = a;
    }

    // Z = 2*log(|S|+eps) + log(scale) + logG[t] - logG[s]; mask only on diagonal tile
    float z[4][8], tmax[8];
#pragma unroll
    for (int r = 0; r < 8; ++r) tmax[r] = -1e30f;
#pragma unroll
    for (int j = 0; j < 4; ++j) {
      const int   kg = kvstart + j * 16 + ln;
      const float gk = ldsGk[j * 16 + ln];
#pragma unroll
      for (int r = 0; r < 8; ++r) {
        float zz = 2.0f * __logf(fabsf(s[j][r]) + EPS) + LOGSCALE + gq[r] - gk;
        if (diag) {
          const int tq = qstart + wave * 16 + half + r;
          if (kg > tq) zz = -1e30f;
        }
        z[j][r] = zz;
        tmax[r] = fmaxf(tmax[r], zz);
      }
    }
    // row-max across 16 lanes (C-layout: lanes are columns)
#pragma unroll
    for (int r = 0; r < 8; ++r)
#pragma unroll
      for (int off = 1; off < 16; off <<= 1)
        tmax[r] = fmaxf(tmax[r], __shfl_xor(tmax[r], off, 32));

    float alpha[8], mnew[8], rsum[8];
#pragma unroll
    for (int r = 0; r < 8; ++r) {
      mnew[r]  = fmaxf(m_run[r], tmax[r]);
      alpha[r] = __expf(m_run[r] - mnew[r]);
      m_run[r] = mnew[r];
      rsum[r]  = 0.f;
    }
#pragma unroll
    for (int j = 0; j < 4; ++j)
#pragma unroll
      for (int r = 0; r < 8; ++r) {
        float p = __expf(z[j][r] - mnew[r]);
        z[j][r] = p;
        rsum[r] += p;
      }
#pragma unroll
    for (int r = 0; r < 8; ++r) {
#pragma unroll
      for (int off = 1; off < 16; off <<= 1)
        rsum[r] += __shfl_xor(rsum[r], off, 32);
      l_run[r] = l_run[r] * alpha[r] + rsum[r];
    }
#pragma unroll
    for (int j = 0; j < 4; ++j)
#pragma unroll
      for (int r = 0; r < 8; ++r) o[j][r] *= alpha[r];

    // C-layout P -> LDS -> A-layout fragments for P@V
#pragma unroll
    for (int j = 0; j < 4; ++j)
#pragma unroll
      for (int r = 0; r < 8; ++r)
        ldsP[wave][(half + r) * LDS_STRIDE + j * 16 + ln] = (bf16)z[j][r];
    __syncthreads();

#pragma unroll
    for (int ks = 0; ks < 2; ++ks) {
      const bf16* prow = &ldsP[wave][ln * LDS_STRIDE + ks * 32];
      v16bf pf = make_frag(prow + half, prow + half + 16);
#pragma unroll
      for (int j = 0; j < 4; ++j) {
        const bf16* vrow = &ldsVt[(j * 16 + ln) * LDS_STRIDE + ks * 32];
        v16bf vf = make_frag(vrow + half, vrow + half + 16);
        o[j] = __builtin_amdgcn_wmma_f32_16x16x32_bf16(
            false, pf, false, vf, (short)0, o[j], false, false);
      }
    }
  }

  // y = O / norm, layout [b][t][h*64+d]
#pragma unroll
  for (int j = 0; j < 4; ++j)
#pragma unroll
    for (int r = 0; r < 8; ++r) {
      int t = qstart + wave * 16 + half + r;
      int c = h * HS + j * 16 + ln;
      out[((size_t)(b * TT + t)) * CC + c] = o[j][r] / l_run[r];
    }
}

extern "C" void kernel_launch(void* const* d_in, const int* in_sizes, int n_in,
                              void* d_out, int out_size, void* d_ws, size_t ws_size,
                              hipStream_t stream) {
  (void)in_sizes; (void)n_in; (void)out_size; (void)ws_size;
  const float* x = (const float*)d_in[0];
  const float* W = (const float*)d_in[1];
  float* out = (float*)d_out;

  char* ws = (char*)d_ws;
  size_t off = 0;
  auto take = [&](size_t bytes) -> void* {
    void* p = ws + off;
    off += (bytes + 255) & ~(size_t)255;
    return p;
  };
  bf16*  Xbf   = (bf16*) take((size_t)BB * TT * CC * 2);
  bf16*  Wbf   = (bf16*) take((size_t)OC * CC * 2);
  bf16*  Qb    = (bf16*) take((size_t)BB * TT * CC * 2);
  bf16*  Kb    = (bf16*) take((size_t)BB * TT * CC * 2);
  bf16*  Vb    = (bf16*) take((size_t)BB * TT * CC * 2);
  float* gates = (float*)take((size_t)BB * TT * NHEAD * 4);
  float* logG  = (float*)take((size_t)BB * NHEAD * TT * 4);

  {
    int n4 = BB * TT * CC / 4;
    cvt_f32_bf16<<<(n4 + 255) / 256, 256, 0, stream>>>(x, Xbf, n4);
  }
  {
    int n4 = OC * CC / 4;
    cvt_f32_bf16<<<(n4 + 255) / 256, 256, 0, stream>>>(W, Wbf, n4);
  }
  qkv_gemm<<<dim3((OC + 63) / 64, BB * TT / 128), 128, 0, stream>>>(Xbf, Wbf, Qb, Kb, Vb, gates);
  gate_scan<<<1, 32, 0, stream>>>(gates, logG);
  attn<<<dim3(TT / 64, BB * NHEAD), 128, 0, stream>>>(Qb, Kb, Vb, logG, out);
}
